// Attention_org_cross_52097953301203
// MI455X (gfx1250) — matrix operand order repaired and checked by reading, not compiled, not run on hardware
//
#include <hip/hip_runtime.h>
#include <hip/hip_bf16.h>

#define BATCH 4
#define NN    4096
#define CC    1024
#define HEADS 4
#define DD    256   // CC / HEADS
#define LDA   40    // padded A-tile leading dim (80 B rows -> 16B-aligned chunks)

typedef __attribute__((ext_vector_type(16))) __bf16 v16bf;
typedef __attribute__((ext_vector_type(8)))  float  v8f;

// ---------------------------------------------------------------------------
// WMMA helper (CDNA5 16x16x32 bf16, f32 accumulate)
// ---------------------------------------------------------------------------
__device__ inline v8f wmma_bf16(v16bf a, v16bf b, v8f c) {
  return __builtin_amdgcn_wmma_f32_16x16x32_bf16(
      /*neg_a=*/false, a, /*neg_b=*/false, b,
      /*c_mod=*/(short)0, c, /*reuse_a=*/false, /*reuse_b=*/false);
}

// A fragment: 16x32 (MxK) from row-major LDS tile, leading dim LDA.
// Lane L: m=L%16, hf=L/16; K = [hf*16 .. +15]. 16B aligned.
__device__ inline v16bf frag_a40(const __bf16* s) {
  int lane = threadIdx.x & 31;
  int m = lane & 15, hf = lane >> 4;
  const uint4* p = (const uint4*)(s + m * LDA + hf * 16);
  const uint4* q = (const uint4*)(s + m * LDA + hf * 16 + 8);
  union { uint4 u[2]; v16bf v; } x;
  x.u[0] = *p; x.u[1] = *q;
  return x.v;
}

// B fragment from swizzled LDS: element (k,n) of the 32x64 B tile lives at
//   Bsw[ ((n>>4)*32 + (k>>4)*16 + (n&15)) * 16 + (k&15) ]
// so lane L of subtile t reads 16 contiguous bf16 (32B, 32B-aligned):
// position within v16bf = (k & 15), owning lane = (k>>4)*16 + (n&15)
// — contiguous k = 16*(L/16) + s, same order as frag_a40 (not the ISA table order).
__device__ inline int bsw_idx(int k, int n) {
  return (((n >> 4) * 32 + (k >> 4) * 16 + (n & 15)) << 4) + (k & 15);
}
__device__ inline v16bf frag_b_sw(const __bf16* sub) {
  int lane = threadIdx.x & 31;
  const uint4* p = (const uint4*)(sub + lane * 16);
  union { uint4 u[2]; v16bf v; } x;
  x.u[0] = p[0]; x.u[1] = p[1];
  return x.v;
}

// Pack 4 f32 -> 4 bf16 (8B)
__device__ inline uint2 pk4(float a, float b, float c, float d) {
  union { uint2 u; __bf16 h[4]; } x;
  x.h[0] = (__bf16)a; x.h[1] = (__bf16)b; x.h[2] = (__bf16)c; x.h[3] = (__bf16)d;
  return x.u;
}

// C/D: f32 16x16 — lane L col n=L%16, VGPR r holds row m = r + 8*(L/16).

// ---------------------------------------------------------------------------
// Kernel 1: per-head projection  P[b,h,n,e] = sum_d X[b,n,h*D+d] * W[h,e,d]
// TR=0: P stored [b,h,n,e] (row-major over n)   — used for V
// TR=1: P stored [b,h,e,n] (transposed)         — used for Q,K (score operands)
// ---------------------------------------------------------------------------
template <bool TR>
__global__ __launch_bounds__(128) void proj_kernel(
    const float* __restrict__ X,   // [B,N,C]
    const float* __restrict__ W,   // [H,D,D] (out,in)
    __bf16* __restrict__ P)
{
  __shared__ __align__(16) __bf16 As[64 * LDA];     // [n][d]
  __shared__ __align__(16) __bf16 Bsw[4 * 32 * 16];
  int bh = blockIdx.z, b = bh / HEADS, h = bh % HEADS;
  int row0 = blockIdx.x * 64;      // n
  int col0 = blockIdx.y * 64;      // e
  const float* Xb = X + (size_t)b * NN * CC + h * DD;
  const float* Wb = W + (size_t)h * DD * DD;
  int tid = threadIdx.x, wave = tid >> 5, lane = tid & 31;
  int rA = tid >> 3, c4 = (tid & 7) * 4;  // A: 16 rows x 8 col-quads
  v8f acc[4] = {};
  for (int k0 = 0; k0 < DD; k0 += 32) {
#pragma unroll
    for (int i = 0; i < 4; ++i) {           // A tile 64(n) x 32(d)
      int r = rA + i * 16;
      float4 f = *(const float4*)(Xb + (size_t)(row0 + r) * CC + k0 + c4);
      *(uint2*)(&As[r * LDA + c4]) = pk4(f.x, f.y, f.z, f.w);
    }
#pragma unroll
    for (int i = 0; i < 4; ++i) {           // B tile 32(k=d) x 64(e)
      int c = rA + i * 16;                  // e-offset
      float4 f = *(const float4*)(Wb + (size_t)(col0 + c) * DD + k0 + c4);
      *(uint2*)(&Bsw[bsw_idx(c4, c)]) = pk4(f.x, f.y, f.z, f.w);
    }
    __syncthreads();
    v16bf a = frag_a40(&As[wave * 16 * LDA]);
#pragma unroll
    for (int t = 0; t < 4; ++t)
      acc[t] = wmma_bf16(a, frag_b_sw(&Bsw[t * 512]), acc[t]);
    __syncthreads();
  }
  __bf16* Pb = P + (size_t)bh * NN * DD;
  if (TR) {
    // [e][n]: each lane's 8 rows (n) are contiguous -> one 16B store per t.
#pragma unroll
    for (int t = 0; t < 4; ++t) {
      int e = col0 + t * 16 + (lane & 15);
      int nb = row0 + wave * 16 + (lane >> 4) * 8;
      union { uint4 u; __bf16 h[8]; } pk;
#pragma unroll
      for (int r = 0; r < 8; ++r) pk.h[r] = (__bf16)acc[t][r];
      *(uint4*)(Pb + (size_t)e * NN + nb) = pk.u;
    }
  } else {
#pragma unroll
    for (int t = 0; t < 4; ++t) {
      int e = col0 + t * 16 + (lane & 15);
#pragma unroll
      for (int r = 0; r < 8; ++r) {
        int n = row0 + wave * 16 + r + (lane >> 4) * 8;
        Pb[(size_t)n * DD + e] = (__bf16)acc[t][r];
      }
    }
  }
}

// ---------------------------------------------------------------------------
// Kernel 2: scores[b,h,i,j] = (1/32) * sum_n Qt[b,h,i,n] * Kt[b,h,j,n]
// Qt/Kt are the transposed projections [b,h,D,N].
// ---------------------------------------------------------------------------
__global__ __launch_bounds__(128) void score_kernel(
    const __bf16* __restrict__ Qt,   // [B,H,D,N]
    const __bf16* __restrict__ Kt,   // [B,H,D,N]
    float* __restrict__ SC)          // [B,H,D,D] f32
{
  __shared__ __align__(16) __bf16 At[64 * LDA];     // [i][n]
  __shared__ __align__(16) __bf16 Bsw[4 * 32 * 16];
  int bh = blockIdx.z;
  const __bf16* Qb = Qt + (size_t)bh * NN * DD;
  const __bf16* Kb = Kt + (size_t)bh * NN * DD;
  int i0 = blockIdx.x * 64, j0 = blockIdx.y * 64;
  int tid = threadIdx.x, wave = tid >> 5, lane = tid & 31;
  int rA = tid >> 3, c4 = (tid & 7) * 4;
  v8f acc[4] = {};
  for (int n0 = 0; n0 < NN; n0 += 32) {
#pragma unroll
    for (int i = 0; i < 4; ++i) {           // A tile 64(i) x 32(n)
      int r = rA + i * 16;
      uint2 v = *(const uint2*)(Qb + (size_t)(i0 + r) * NN + n0 + c4);
      *(uint2*)(&At[r * LDA + c4]) = v;
    }
#pragma unroll
    for (int i = 0; i < 4; ++i) {           // B tile 32(k=n) x 64(j)
      int c = rA + i * 16;                  // j-offset
      uint2 v = *(const uint2*)(Kb + (size_t)(j0 + c) * NN + n0 + c4);
      *(uint2*)(&Bsw[bsw_idx(c4, c)]) = v;
    }
    __syncthreads();
    v16bf a = frag_a40(&At[wave * 16 * LDA]);
#pragma unroll
    for (int t = 0; t < 4; ++t)
      acc[t] = wmma_bf16(a, frag_b_sw(&Bsw[t * 512]), acc[t]);
    __syncthreads();
  }
  const float scale = 0.03125f;              // 1/sqrt(1024)
  float* out = SC + (size_t)bh * DD * DD;
#pragma unroll
  for (int t = 0; t < 4; ++t) {
    int j = j0 + t * 16 + (lane & 15);
#pragma unroll
    for (int r = 0; r < 8; ++r) {
      int i = i0 + wave * 16 + r + (lane >> 4) * 8;
      out[(size_t)i * DD + j] = acc[t][r] * scale;
    }
  }
}

// ---------------------------------------------------------------------------
// Kernel 3: InstanceNorm over (D,D) per (b,h) + row softmax -> bf16 probs
// ---------------------------------------------------------------------------
__global__ __launch_bounds__(256) void normsoftmax_kernel(
    const float* __restrict__ SC, __bf16* __restrict__ PR)
{
  __shared__ float red[256], red2[256];
  __shared__ float stats[2];
  int bh = blockIdx.x;
  const float* s = SC + (size_t)bh * DD * DD;
  int tid = threadIdx.x;
  float sum = 0.f, sq = 0.f;
#pragma unroll 4
  for (int i = tid * 4; i < DD * DD; i += 1024) {
    float4 v = *(const float4*)(s + i);
    sum += v.x + v.y + v.z + v.w;
    sq += v.x * v.x + v.y * v.y + v.z * v.z + v.w * v.w;
  }
  red[tid] = sum; red2[tid] = sq;
  __syncthreads();
  for (int off = 128; off > 0; off >>= 1) {
    if (tid < off) { red[tid] += red[tid + off]; red2[tid] += red2[tid + off]; }
    __syncthreads();
  }
  if (tid == 0) {
    float m = red[0] * (1.f / (DD * DD));
    float var = red2[0] * (1.f / (DD * DD)) - m * m;
    stats[0] = m; stats[1] = rsqrtf(var + 1e-5f);
  }
  __syncthreads();
  float mean = stats[0], rstd = stats[1];
  int wave = tid >> 5, lane = tid & 31;
  __bf16* p = PR + (size_t)bh * DD * DD;
  for (int row = wave; row < DD; row += 8) {
    const float* rs = s + (size_t)row * DD;
    float z[8], mx = -3.4e38f;
#pragma unroll
    for (int q = 0; q < 8; ++q) {
      z[q] = (rs[lane + 32 * q] - mean) * rstd;
      mx = fmaxf(mx, z[q]);
    }
#pragma unroll
    for (int m2 = 16; m2 > 0; m2 >>= 1) mx = fmaxf(mx, __shfl_xor(mx, m2, 32));
    float se = 0.f;
#pragma unroll
    for (int q = 0; q < 8; ++q) { z[q] = __expf(z[q] - mx); se += z[q]; }
#pragma unroll
    for (int m2 = 16; m2 > 0; m2 >>= 1) se += __shfl_xor(se, m2, 32);
    float inv = 1.f / se;
#pragma unroll
    for (int q = 0; q < 8; ++q)
      p[(size_t)row * DD + lane + 32 * q] = (__bf16)(z[q] * inv);
  }
}

// ---------------------------------------------------------------------------
// Kernel 4: ctx[b,h,i,n] = sum_j probs[b,h,i,j] * V[b,h,n,j], scattered into
// CTX[b, n, i*H + h] (permute(0,3,2,1) + head-fastest reshape), bf16.
// ---------------------------------------------------------------------------
__global__ __launch_bounds__(128) void ctx_kernel(
    const __bf16* __restrict__ PRb,  // [B,H,D,D]
    const __bf16* __restrict__ Vp,   // [B,H,N,D]
    __bf16* __restrict__ CTX)        // [B,N,C]
{
  __shared__ __align__(16) __bf16 As[64 * LDA];     // [i][j]
  __shared__ __align__(16) __bf16 Bsw[4 * 32 * 16];
  int bh = blockIdx.z, b = bh / HEADS, h = bh % HEADS;
  const __bf16* Pb = PRb + (size_t)bh * DD * DD;
  const __bf16* Vb = Vp + (size_t)bh * NN * DD;
  int i0 = blockIdx.x * 64, n0 = blockIdx.y * 64;
  int tid = threadIdx.x, wave = tid >> 5, lane = tid & 31;
  int rA = tid >> 3, c4 = (tid & 7) * 4;
  v8f acc[4] = {};
  for (int k0 = 0; k0 < DD; k0 += 32) {
#pragma unroll
    for (int i = 0; i < 4; ++i) {           // A tile 64(i) x 32(j)
      int r = rA + i * 16;
      uint2 v = *(const uint2*)(Pb + (size_t)(i0 + r) * DD + k0 + c4);
      *(uint2*)(&As[r * LDA + c4]) = v;
    }
#pragma unroll
    for (int i = 0; i < 4; ++i) {           // B tile 32(k=j) x 64(n)
      int c = rA + i * 16;                  // n-offset
      uint2 v = *(const uint2*)(Vb + (size_t)(n0 + c) * DD + k0 + c4);
      *(uint2*)(&Bsw[bsw_idx(c4, c)]) = v;
    }
    __syncthreads();
    v16bf a = frag_a40(&As[wave * 16 * LDA]);
#pragma unroll
    for (int t = 0; t < 4; ++t)
      acc[t] = wmma_bf16(a, frag_b_sw(&Bsw[t * 512]), acc[t]);
    __syncthreads();
  }
#pragma unroll
  for (int t = 0; t < 4; ++t) {
    int n = n0 + t * 16 + (lane & 15);
#pragma unroll
    for (int r = 0; r < 8; ++r) {
      int i = i0 + wave * 16 + r + (lane >> 4) * 8;
      CTX[((size_t)b * NN + n) * CC + i * HEADS + h] = (__bf16)acc[t][r];
    }
  }
}

// ---------------------------------------------------------------------------
// Kernel 5: out[b,n,o] = sum_c CTX[b,n,c] * Wout[o,c]   (rows = b*N+n flat)
// ---------------------------------------------------------------------------
__global__ __launch_bounds__(128) void outproj_kernel(
    const __bf16* __restrict__ CTX,  // [B*N, C] bf16
    const float* __restrict__ Wo,    // [C, C] (out,in)
    float* __restrict__ Out)         // [B*N, C] f32
{
  __shared__ __align__(16) __bf16 As[64 * LDA];     // [row][c]
  __shared__ __align__(16) __bf16 Bsw[4 * 32 * 16];
  int r0 = blockIdx.x * 64, c0 = blockIdx.y * 64;
  int tid = threadIdx.x, wave = tid >> 5, lane = tid & 31;
  int rA = tid >> 3, c4 = (tid & 7) * 4;
  v8f acc[4] = {};
  for (int k0 = 0; k0 < CC; k0 += 32) {
#pragma unroll
    for (int i = 0; i < 4; ++i) {           // A tile 64(row) x 32(c)
      int r = rA + i * 16;
      uint2 v = *(const uint2*)(CTX + (size_t)(r0 + r) * CC + k0 + c4);
      *(uint2*)(&As[r * LDA + c4]) = v;
    }
#pragma unroll
    for (int i = 0; i < 4; ++i) {           // B tile 32(k=c) x 64(o)
      int c = rA + i * 16;                  // o-offset
      float4 f = *(const float4*)(Wo + (size_t)(c0 + c) * CC + k0 + c4);
      *(uint2*)(&Bsw[bsw_idx(c4, c)]) = pk4(f.x, f.y, f.z, f.w);
    }
    __syncthreads();
    v16bf a = frag_a40(&As[wave * 16 * LDA]);
#pragma unroll
    for (int t = 0; t < 4; ++t)
      acc[t] = wmma_bf16(a, frag_b_sw(&Bsw[t * 512]), acc[t]);
    __syncthreads();
  }
#pragma unroll
  for (int t = 0; t < 4; ++t) {
    int o = c0 + t * 16 + (lane & 15);
#pragma unroll
    for (int r = 0; r < 8; ++r) {
      int row = r0 + wave * 16 + r + (lane >> 4) * 8;
      Out[(size_t)row * CC + o] = acc[t][r];
    }
  }
}

// ---------------------------------------------------------------------------
extern "C" void kernel_launch(void* const* d_in, const int* in_sizes, int n_in,
                              void* d_out, int out_size, void* d_ws, size_t ws_size,
                              hipStream_t stream) {
  const float* S    = (const float*)d_in[0];
  const float* SKV  = (const float*)d_in[1];
  const float* T    = (const float*)d_in[2];
  const float* TKV  = (const float*)d_in[3];
  const float* Wq   = (const float*)d_in[4];
  const float* Wk   = (const float*)d_in[5];
  const float* Wv   = (const float*)d_in[6];
  const float* Wout = (const float*)d_in[7];
  const float* Woutd= (const float*)d_in[8];
  float* out = (float*)d_out;

  // Workspace (~241 MB): Qt,Kt,Qdt,Kdt transposed [B,H,D,N]; V,Vd [B,H,N,D];
  // scores f32 + probs bf16 + merged ctx bf16 (reused per attend).
  char* ws = (char*)d_ws;
  const size_t NE = (size_t)BATCH * HEADS * NN * DD;
  __bf16* Qt  = (__bf16*)ws; ws += NE * sizeof(__bf16);
  __bf16* Kt  = (__bf16*)ws; ws += NE * sizeof(__bf16);
  __bf16* V   = (__bf16*)ws; ws += NE * sizeof(__bf16);
  __bf16* Qdt = (__bf16*)ws; ws += NE * sizeof(__bf16);
  __bf16* Kdt = (__bf16*)ws; ws += NE * sizeof(__bf16);
  __bf16* Vd  = (__bf16*)ws; ws += NE * sizeof(__bf16);
  float*  SCb = (float*)ws;  ws += (size_t)BATCH * HEADS * DD * DD * sizeof(float);
  __bf16* PRb = (__bf16*)ws; ws += (size_t)BATCH * HEADS * DD * DD * sizeof(__bf16);
  __bf16* CTX = (__bf16*)ws; ws += (size_t)BATCH * NN * CC * sizeof(__bf16);

  dim3 blk(128);
  dim3 gproj(NN / 64, DD / 64, BATCH * HEADS);
  proj_kernel<true ><<<gproj, blk, 0, stream>>>(S,   Wq, Qt);
  proj_kernel<true ><<<gproj, blk, 0, stream>>>(SKV, Wk, Kt);
  proj_kernel<false><<<gproj, blk, 0, stream>>>(SKV, Wv, V);
  proj_kernel<true ><<<gproj, blk, 0, stream>>>(T,   Wq, Qdt);
  proj_kernel<true ><<<gproj, blk, 0, stream>>>(TKV, Wk, Kdt);
  proj_kernel<false><<<gproj, blk, 0, stream>>>(TKV, Wv, Vd);

  // Output tuple order: (S_out, T2S, S2T, T_out)
  const __bf16* Qs[4] = {Qt,  Qdt, Qt,  Qdt};
  const __bf16* Ks[4] = {Kt,  Kt,  Kdt, Kdt};
  const __bf16* Vs[4] = {V,   V,   Vd,  Vd};
  const float*  Ws[4] = {Wout, Woutd, Wout, Woutd};

  for (int a = 0; a < 4; ++a) {
    score_kernel<<<dim3(DD / 64, DD / 64, BATCH * HEADS), blk, 0, stream>>>(
        Qs[a], Ks[a], SCb);
    normsoftmax_kernel<<<dim3(BATCH * HEADS), dim3(256), 0, stream>>>(SCb, PRb);
    ctx_kernel<<<dim3(DD / 64, NN / 64, BATCH * HEADS), blk, 0, stream>>>(
        PRb, Vs[a], CTX);
    outproj_kernel<<<dim3((BATCH * NN) / 64, CC / 64), blk, 0, stream>>>(
        CTX, Ws[a], out + (size_t)a * BATCH * NN * CC);
  }
}
